// MultiHeadSelfAttentionModule_25898652795599
// MI455X (gfx1250) — matrix operand order repaired and checked by reading, not compiled, not run on hardware
//
#include <hip/hip_runtime.h>
#include <hip/hip_bf16.h>
#include <cstdint>

#define DMODEL 512
#define NHEADS 8
#define DK     64
#define TLEN   2048
#define BATCH  4
#define ROWS   (BATCH * TLEN)    // 8192
#define WELEM  (DMODEL * DMODEL) // 262144

typedef __bf16 bf16;
typedef __attribute__((ext_vector_type(16))) __bf16 bf16x16;
typedef __attribute__((ext_vector_type(8)))  __bf16 bf16x8;
typedef __attribute__((ext_vector_type(8)))  float  f32x8;
typedef __attribute__((ext_vector_type(4)))  float  f32x4;
typedef __attribute__((ext_vector_type(4)))  unsigned u32x4;
typedef __attribute__((ext_vector_type(8)))  int      i32x8;
typedef __attribute__((ext_vector_type(4)))  int      i32x4;

union BF16Frag { bf16x16 v; bf16x8 h[2]; };

static __device__ __forceinline__ f32x8 wmma_bf16(const BF16Frag& a, const BF16Frag& b, f32x8 c) {
  // D = A(16x32 bf16) x B(32x16 bf16) + C(16x16 f32)
  return __builtin_amdgcn_wmma_f32_16x16x32_bf16(false, a.v, false, b.v, (short)0, c, false, false);
}

// Flat LDS pointers carry the LDS byte address in their low 32 bits
// (ISA 10.2: LDS aperture maps addr[31:0] -> LDS_ADDR).
static __device__ __forceinline__ unsigned lds_addr(const void* p) {
  return (unsigned)(uintptr_t)p;
}

// ASYNCcnt-tracked copy of 16 bytes global -> LDS, no VGPR data movement.
static __device__ __forceinline__ void async_copy_b128(unsigned ldsDst, const void* gsrc) {
  asm volatile("global_load_async_to_lds_b128 %0, %1, off"
               :: "v"(ldsDst), "v"((unsigned long long)(uintptr_t)gsrc)
               : "memory");
}
static __device__ __forceinline__ void wait_asynccnt0() {
  asm volatile("s_wait_asynccnt 0x0" ::: "memory");
}

// ---------------------------------------------------------------------------
// 1) LayerNorm: one wave per 512-elem row, fp32 stats, bf16 output
// ---------------------------------------------------------------------------
__global__ void __launch_bounds__(256) ln_kernel(const float* __restrict__ x,
                                                 const float* __restrict__ g,
                                                 const float* __restrict__ bta,
                                                 bf16* __restrict__ normed) {
  int wave = threadIdx.x >> 5, lane = threadIdx.x & 31;
  size_t row = (size_t)blockIdx.x * 8 + wave;
  const float* xr = x + row * DMODEL;
  f32x4 v[4];
  float s = 0.f, ss = 0.f;
#pragma unroll
  for (int i = 0; i < 4; ++i) {
    v[i] = *(const f32x4*)&xr[lane * 16 + i * 4];
#pragma unroll
    for (int j = 0; j < 4; ++j) { s += v[i][j]; ss += v[i][j] * v[i][j]; }
  }
#pragma unroll
  for (int off = 16; off >= 1; off >>= 1) {
    s  += __shfl_xor(s,  off, 32);
    ss += __shfl_xor(ss, off, 32);
  }
  float mu   = s  * (1.f / DMODEL);
  float var  = ss * (1.f / DMODEL) - mu * mu;
  float rstd = rsqrtf(var + 1e-5f);
  bf16* nr = normed + row * DMODEL;
#pragma unroll
  for (int i = 0; i < 4; ++i)
#pragma unroll
    for (int j = 0; j < 4; ++j) {
      int d = lane * 16 + i * 4 + j;
      nr[d] = (bf16)((v[i][j] - mu) * rstd * g[d] + bta[d]);
    }
}

// ---------------------------------------------------------------------------
// 2) Weight convert + transpose: wt[m][d*512+o] = w_m[o*512+d]  (bf16)
// ---------------------------------------------------------------------------
__global__ void __launch_bounds__(256) wtr_kernel(const float* __restrict__ wq,
                                                  const float* __restrict__ wk,
                                                  const float* __restrict__ wv,
                                                  const float* __restrict__ wo,
                                                  bf16* __restrict__ wt) {
  int idx = blockIdx.x * 256 + threadIdx.x;  // 0 .. 4*262144-1
  int m = idx >> 18;
  int e = idx & (WELEM - 1);
  int d = e >> 9, o = e & 511;
  const float* w = (m == 0) ? wq : (m == 1) ? wk : (m == 2) ? wv : wo;
  wt[(size_t)m * WELEM + e] = (bf16)w[(size_t)o * DMODEL + d];
}

// ---------------------------------------------------------------------------
// 3) Fused QKV GEMM.  A-tile staged to LDS via async global->LDS copies.
// ---------------------------------------------------------------------------
__global__ void __launch_bounds__(256) qkv_kernel(const bf16* __restrict__ normed,
                                                  const bf16* __restrict__ wt,
                                                  const float* __restrict__ bq,
                                                  const float* __restrict__ bk,
                                                  const float* __restrict__ bv,
                                                  bf16* __restrict__ q,
                                                  bf16* __restrict__ k,
                                                  bf16* __restrict__ v) {
  __shared__ bf16 aLds[16 * DMODEL];  // 16 KB: 16 input rows
  int tid = threadIdx.x, wave = tid >> 5, lane = tid & 31;
  int row0 = blockIdx.x * 16;
  {
    int r  = tid >> 4;
    int c0 = (tid & 15) * 32;
    const bf16* src = normed + (size_t)(row0 + r) * DMODEL + c0;
    unsigned    dst = lds_addr(&aLds[r * DMODEL + c0]);
#pragma unroll
    for (int i = 0; i < 4; ++i)
      async_copy_b128(dst + i * 16, src + i * 8);
    wait_asynccnt0();
  }
  __syncthreads();

  int m = lane & 15, hi = lane >> 4;
  int bIdx  = row0 >> 11;            // batch
  int tBase = row0 & (TLEN - 1);     // time within batch (16 | 2048)
  const bf16*  wts[3]  = { wt, wt + WELEM, wt + 2 * (size_t)WELEM };
  const float* bias[3] = { bq, bk, bv };
  bf16*        outs[3] = { q, k, v };

  for (int mat = 0; mat < 3; ++mat) {
    const bf16* wtm = wts[mat];
#pragma unroll
    for (int ct = 0; ct < 4; ++ct) {
      int n0 = (wave + ct * 8) * 16;
      f32x8 acc = {};
      for (int kc = 0; kc < DMODEL; kc += 32) {
        BF16Frag a, b;
        a.h[0] = *(const bf16x8*)&aLds[m * DMODEL + kc + hi * 8];
        a.h[1] = *(const bf16x8*)&aLds[m * DMODEL + kc + 16 + hi * 8];
#pragma unroll
        for (int s = 0; s < 16; ++s)
          b.v[s] = wtm[(size_t)(kc + (s & 8) * 2 + 8 * hi + (s & 7)) * DMODEL + n0 + m];
        acc = wmma_bf16(a, b, acc);
      }
      int n = n0 + m;                 // output feature for this lane
      float bb = bias[mat][n];
      int h = n >> 6, c = n & 63;
      bf16* op = outs[mat] + ((size_t)(bIdx * NHEADS + h) * TLEN) * DK + c;
#pragma unroll
      for (int r = 0; r < 8; ++r) {
        int t = tBase + r + 8 * hi;
        float val = acc[r] + bb;
        if (mat == 0) val *= 0.125f;  // fold 1/sqrt(64) into Q
        if (mat == 1) {               // additive sinusoidal PE on K
          int i2 = c & ~1;            // 2*i
          float ang = (float)t * __expf(-0.14391156f * (float)i2); // ln(1e4)/64
          val += (c & 1) ? __cosf(ang) : __sinf(ang);
        }
        op[(size_t)t * DK] = (bf16)val;
      }
    }
  }
}

// ---------------------------------------------------------------------------
// 4) Flash attention.  V tile DMA'd by the Tensor Data Mover (wave 0 issues,
//    TENSORcnt wait), K tile transposed through VGPRs, P->bf16 transpose via
//    per-wave LDS slab.  Online softmax in fp32.
// ---------------------------------------------------------------------------
__global__ void __launch_bounds__(256) attn_kernel(const bf16* __restrict__ q,
                                                   const bf16* __restrict__ k,
                                                   const bf16* __restrict__ v,
                                                   bf16* __restrict__ ctx) {
  __shared__ bf16 KtLds[64 * 32];      // K^T tile: [c][s_local]
  __shared__ bf16 VLds[32 * 64];       // V tile:   [s_local][c]
  __shared__ bf16 PLds[8][16 * 32];    // per-wave P transpose slab
  int tid = threadIdx.x, wave = tid >> 5, lane = tid & 31;
  int m = lane & 15, hi = lane >> 4;
  int bh = blockIdx.y;
  int t0 = blockIdx.x * 128 + wave * 16;
  const bf16* qb = q + (size_t)bh * TLEN * DK;
  const bf16* kb = k + (size_t)bh * TLEN * DK;
  const bf16* vb = v + (size_t)bh * TLEN * DK;

  // Q A-fragments for the two 32-wide K chunks of dk=64
  BF16Frag aq[2];
#pragma unroll
  for (int kc = 0; kc < 2; ++kc) {
    const bf16* qp = qb + (size_t)(t0 + m) * DK + kc * 32 + hi * 8;
    aq[kc].h[0] = *(const bf16x8*)qp;
    aq[kc].h[1] = *(const bf16x8*)(qp + 16);
  }

  f32x8 accv[4] = {};                  // 16x64 f32 context accumulator
  float mrow[8], lrow[8];
#pragma unroll
  for (int r = 0; r < 8; ++r) { mrow[r] = -1e30f; lrow[r] = 0.f; }

  for (int s0 = 0; s0 < TLEN; s0 += 32) {
    // ---- stage V tile: one TDM descriptor, issued by wave 0 only ----
#if __has_builtin(__builtin_amdgcn_tensor_load_to_lds)
    if (wave == 0) {
      unsigned long long ga = (unsigned long long)(uintptr_t)(vb + (size_t)s0 * DK);
      u32x4 g0; i32x8 g1; i32x4 g2 = {0, 0, 0, 0}, g3 = {0, 0, 0, 0};
      i32x8 g4 = {0, 0, 0, 0, 0, 0, 0, 0};
      g0[0] = 1u;                                   // count=1, user mode
      g0[1] = lds_addr(&VLds[0]);                   // lds_addr
      g0[2] = (unsigned)(ga & 0xffffffffu);         // global_addr[31:0]
      g0[3] = (unsigned)((ga >> 32) & 0x01ffffffu)  // global_addr[56:32]
            | (2u << 30);                           // type = 2 ("image")
      g1[0] = (int)(1u << 16);                      // wg_mask=0, data_size=1 (2B)
      g1[1] = (int)((64u & 0xffffu) << 16);         // tensor_dim0 = 64 (lo16)
      g1[2] = (int)((2048u & 0xffffu) << 16);       // dim0 hi16=0 | tensor_dim1=2048 lo16
      g1[3] = (int)(64u << 16);                     // dim1 hi16=0 | tile_dim0=64
      g1[4] = 32;                                   // tile_dim1=32, tile_dim2=0
      g1[5] = 64;                                   // tensor_dim0_stride = 64
      g1[6] = 0; g1[7] = 0;                         // dim1_stride unused (2D tile)
      __builtin_amdgcn_tensor_load_to_lds(g0, g1, g2, g3, g4, 0);
      __builtin_amdgcn_s_wait_tensorcnt((short)0);
    }
#else
    {
      int j  = tid >> 3;
      int c0 = (tid & 7) * 8;
      async_copy_b128(lds_addr(&VLds[j * DK + c0]),
                      &vb[(size_t)(s0 + j) * DK + c0]);
      wait_asynccnt0();
    }
#endif
    // ---- stage K^T tile (transpose through VGPRs) ----
    {
      int j  = tid >> 3;               // 0..31  (s_local)
      int c0 = (tid & 7) * 8;          // 0..56
      bf16x8 krow = *(const bf16x8*)&kb[(size_t)(s0 + j) * DK + c0];
#pragma unroll
      for (int u = 0; u < 8; ++u) KtLds[(c0 + u) * 32 + j] = krow[u];
      if (s0 + 32 < TLEN) {            // prefetch next tiles (global_prefetch_b8)
        __builtin_prefetch(&kb[(size_t)(s0 + 32 + j) * DK + c0], 0, 1);
        __builtin_prefetch(&vb[(size_t)(s0 + 32 + j) * DK + c0], 0, 1);
      }
    }
    __syncthreads();

    // scores: S0 = Q x K[s0..s0+15]^T, S1 = Q x K[s0+16..s0+31]^T
    f32x8 sc0 = {}, sc1 = {};
#pragma unroll
    for (int kc = 0; kc < 2; ++kc) {
      BF16Frag b0, b1;
#pragma unroll
      for (int s = 0; s < 16; ++s) {
        const bf16* kp = &KtLds[(kc * 32 + (s & 8) * 2 + 8 * hi + (s & 7)) * 32];
        b0.v[s] = kp[m]; b1.v[s] = kp[16 + m];
      }
      sc0 = wmma_bf16(aq[kc], b0, sc0);
      sc1 = wmma_bf16(aq[kc], b1, sc1);
    }

    // online softmax: row stats live across the 16-lane group of each half
    float mx[8];
#pragma unroll
    for (int r = 0; r < 8; ++r) mx[r] = fmaxf(sc0[r], sc1[r]);
#pragma unroll
    for (int off = 8; off >= 1; off >>= 1)
#pragma unroll
      for (int r = 0; r < 8; ++r) mx[r] = fmaxf(mx[r], __shfl_xor(mx[r], off, 32));
    float cr[8], rs[8];
#pragma unroll
    for (int r = 0; r < 8; ++r) {
      float mn = fmaxf(mrow[r], mx[r]);
      cr[r] = __expf(mrow[r] - mn);
      mrow[r] = mn;
      float p0 = __expf(sc0[r] - mn);
      float p1 = __expf(sc1[r] - mn);
      sc0[r] = p0; sc1[r] = p1;
      rs[r] = p0 + p1;
    }
#pragma unroll
    for (int off = 8; off >= 1; off >>= 1)
#pragma unroll
      for (int r = 0; r < 8; ++r) rs[r] += __shfl_xor(rs[r], off, 32);
#pragma unroll
    for (int r = 0; r < 8; ++r) lrow[r] = lrow[r] * cr[r] + rs[r];
#pragma unroll
    for (int jj = 0; jj < 4; ++jj)
#pragma unroll
      for (int r = 0; r < 8; ++r) accv[jj][r] *= cr[r];

    // P (f32 C-layout) -> bf16 A-layout via per-wave LDS slab
    bf16* pw = &PLds[wave][0];
#pragma unroll
    for (int r = 0; r < 8; ++r) {
      pw[(r + 8 * hi) * 32 + m]      = (bf16)sc0[r];
      pw[(r + 8 * hi) * 32 + 16 + m] = (bf16)sc1[r];
    }
    BF16Frag pa;   // same-wave DS ordering guarantees write->read
    pa.h[0] = *(const bf16x8*)&pw[m * 32 + hi * 8];
    pa.h[1] = *(const bf16x8*)&pw[m * 32 + 16 + hi * 8];

    // context += P(16x32) x V(32x64)
#pragma unroll
    for (int jj = 0; jj < 4; ++jj) {
      BF16Frag bvf;
#pragma unroll
      for (int s = 0; s < 16; ++s)
        bvf.v[s] = VLds[((s & 8) * 2 + 8 * hi + (s & 7)) * DK + jj * 16 + m];
      accv[jj] = wmma_bf16(pa, bvf, accv[jj]);
    }
    __syncthreads();
  }

  // normalize and store bf16 context in [B][T][D] layout for the out-proj
  int b = bh >> 3, h = bh & 7;
#pragma unroll
  for (int r = 0; r < 8; ++r) {
    float inv = 1.f / lrow[r];
    int t = t0 + r + 8 * hi;
    bf16* cp = ctx + ((size_t)(b * TLEN + t)) * DMODEL + h * DK;
#pragma unroll
    for (int jj = 0; jj < 4; ++jj)
      cp[jj * 16 + m] = (bf16)(accv[jj][r] * inv);
  }
}

// ---------------------------------------------------------------------------
// 5) Output projection: out = ctx @ wo^T + bo   (f32 output)
// ---------------------------------------------------------------------------
__global__ void __launch_bounds__(256) proj_kernel(const bf16* __restrict__ ctx,
                                                   const bf16* __restrict__ wto,
                                                   const float* __restrict__ bo,
                                                   float* __restrict__ out) {
  __shared__ bf16 aLds[16 * DMODEL];
  int tid = threadIdx.x, wave = tid >> 5, lane = tid & 31;
  int row0 = blockIdx.x * 16;
  {
    int r  = tid >> 4;
    int c0 = (tid & 15) * 32;
    const bf16* src = ctx + (size_t)(row0 + r) * DMODEL + c0;
    unsigned    dst = lds_addr(&aLds[r * DMODEL + c0]);
#pragma unroll
    for (int i = 0; i < 4; ++i)
      async_copy_b128(dst + i * 16, src + i * 8);
    wait_asynccnt0();
  }
  __syncthreads();

  int m = lane & 15, hi = lane >> 4;
#pragma unroll
  for (int ct = 0; ct < 4; ++ct) {
    int n0 = (wave + ct * 8) * 16;
    f32x8 acc = {};
    for (int kc = 0; kc < DMODEL; kc += 32) {
      BF16Frag a, b;
      a.h[0] = *(const bf16x8*)&aLds[m * DMODEL + kc + hi * 8];
      a.h[1] = *(const bf16x8*)&aLds[m * DMODEL + kc + 16 + hi * 8];
#pragma unroll
      for (int s = 0; s < 16; ++s)
        b.v[s] = wto[(size_t)(kc + (s & 8) * 2 + 8 * hi + (s & 7)) * DMODEL + n0 + m];
      acc = wmma_bf16(a, b, acc);
    }
    int n = n0 + m;
    float bb = bo[n];
#pragma unroll
    for (int r = 0; r < 8; ++r)
      out[(size_t)(row0 + r + 8 * hi) * DMODEL + n] = acc[r] + bb;
  }
}

// ---------------------------------------------------------------------------
extern "C" void kernel_launch(void* const* d_in, const int* in_sizes, int n_in,
                              void* d_out, int out_size, void* d_ws, size_t ws_size,
                              hipStream_t stream) {
  const float* x    = (const float*)d_in[0];
  const float* ln_g = (const float*)d_in[1];
  const float* ln_b = (const float*)d_in[2];
  const float* wq   = (const float*)d_in[3];
  const float* bq   = (const float*)d_in[4];
  const float* wk   = (const float*)d_in[5];
  const float* bk   = (const float*)d_in[6];
  const float* wv   = (const float*)d_in[7];
  const float* bv   = (const float*)d_in[8];
  const float* wo   = (const float*)d_in[9];
  const float* bo   = (const float*)d_in[10];

  char* ws = (char*)d_ws;
  const size_t MB = 1024 * 1024;
  bf16* normed = (bf16*)(ws);                 // 8 MB  (8192 x 512 bf16)
  bf16* wt     = (bf16*)(ws + 8  * MB);       // 2 MB  (4 x 512 x 512 bf16, transposed)
  bf16* qws    = (bf16*)(ws + 10 * MB);       // 8 MB  [B][H][T][64]
  bf16* kws    = (bf16*)(ws + 18 * MB);       // 8 MB
  bf16* vws    = (bf16*)(ws + 26 * MB);       // 8 MB
  bf16* ctx    = (bf16*)(ws + 34 * MB);       // 8 MB  [B*T][D]

  ln_kernel  <<<ROWS / 8, 256, 0, stream>>>(x, ln_g, ln_b, normed);
  wtr_kernel <<<(4 * WELEM) / 256, 256, 0, stream>>>(wq, wk, wv, wo, wt);
  qkv_kernel <<<ROWS / 16, 256, 0, stream>>>(normed, wt, bq, bk, bv, qws, kws, vws);
  attn_kernel<<<dim3(TLEN / 128, BATCH * NHEADS), 256, 0, stream>>>(qws, kws, vws, ctx);
  proj_kernel<<<ROWS / 16, 256, 0, stream>>>(ctx, wt + 3 * (size_t)WELEM, bo, (float*)d_out);
}
